// MatMulTransform_7705171329721
// MI455X (gfx1250) — matrix operand order repaired and checked by reading, not compiled, not run on hardware
//
#include <hip/hip_runtime.h>

#define B_ 8
#define L_ 4096
#define D_ 768

typedef __attribute__((ext_vector_type(16))) _Float16 v16h;
typedef __attribute__((ext_vector_type(8)))  float    v8f;

// ---------------------------------------------------------------------------
// Kernel 0: zero the atomic accumulators (meanSum[B*D] and msum[B], contiguous)
// ---------------------------------------------------------------------------
__global__ void zero_ws_kernel(float* __restrict__ p, int n) {
    for (int i = threadIdx.x; i < n; i += blockDim.x) p[i] = 0.0f;
}

// ---------------------------------------------------------------------------
// Kernel 1: column sums over L.  meanSum[b,d] = sum_l x[b,l,d]
// Block = 192 threads; thread t owns float4 columns d = 4t..4t+3.
// Grid  = B * 64 blocks; each block reduces 64 rows, then 4 f32 atomics/thread.
// Loads are fully coalesced (each row iteration reads one contiguous 3 KB row).
// ---------------------------------------------------------------------------
__global__ void __launch_bounds__(192)
colsum_kernel(const float* __restrict__ x, float* __restrict__ meanSum) {
    const int t  = threadIdx.x;              // 0..191
    const int b  = blockIdx.x >> 6;          // /64
    const int lc = blockIdx.x & 63;          // %64
    const float* xp = x + (size_t)b * L_ * D_ + (size_t)lc * 64 * D_ + 4 * t;
    float4 s = make_float4(0.f, 0.f, 0.f, 0.f);
    #pragma unroll 4
    for (int r = 0; r < 64; ++r) {
        float4 v = *reinterpret_cast<const float4*>(xp + (size_t)r * D_);
        s.x += v.x; s.y += v.y; s.z += v.z; s.w += v.w;
    }
    float* mp = meanSum + b * D_ + 4 * t;
    unsafeAtomicAdd(mp + 0, s.x);
    unsafeAtomicAdd(mp + 1, s.y);
    unsafeAtomicAdd(mp + 2, s.z);
    unsafeAtomicAdd(mp + 3, s.w);
}

// ---------------------------------------------------------------------------
// Kernel 2 (WMMA): y1[b,l] = (1/D) * dot(x[b,l,:], meanSum[b,:]/L)
// Wave-per-16-row-tile; K looped in steps of 32 using v_wmma_f32_16x16x32_f16.
//   A layout (16-bit 16x32): lane L -> row M=L%16, K = 8*(L/16)+{0..7} in e0..7,
//                            K = 16+8*(L/16)+{0..7} in e8..15.
//   B layout (32x16): lane group g=L/16 holds K = 8g+{0..7} in e0..7, 16+8g+{0..7} in e8..15.
//   B is constant across N (all 16 columns = mean vector) so every column of
//   the accumulator holds the dot product; lanes 0 and 16 extract rows
//   (C/D layout: VGPR r holds M=r for lanes 0-15, M=r+8 for lanes 16-31).
// Also accumulates msum[b] = sum_l y1[b,l] via f32 atomic.
// ---------------------------------------------------------------------------
__global__ void __launch_bounds__(128)
dot_wmma_kernel(const float* __restrict__ x, const float* __restrict__ meanSum,
                float* __restrict__ y1, float* __restrict__ msum) {
    const int b    = blockIdx.y;
    const int wave = threadIdx.x >> 5;       // 0..3
    const int lane = threadIdx.x & 31;
    const int g    = lane >> 4;              // lane group: 0 or 1
    const int m    = lane & 15;              // row within tile
    const int l0   = blockIdx.x * 64 + wave * 16;

    const float* xrow = x + ((size_t)b * L_ + (l0 + m)) * D_;
    const float* ms   = meanSum + b * D_;
    const float  invL = 1.0f / (float)L_;

    v8f acc = {};
    #pragma unroll 4
    for (int kb = 0; kb < D_; kb += 32) {
        // A operand: 8 floats at d = kb + 8g, and 8 floats at d = kb + 8g + 16
        const float4* xa = reinterpret_cast<const float4*>(xrow + kb + 8 * g);
        float4 p0 = xa[0], p1 = xa[1];       // K = 8g + 0..7
        float4 p2 = xa[4], p3 = xa[5];       // K = 8g + 16..23
        // B operand: mean values at d = kb + 8g + 0..7 and kb + 8g + 16..23 (scaled by 1/L)
        const float4* mb = reinterpret_cast<const float4*>(ms + kb + 8 * g);
        float4 q0 = mb[0], q1 = mb[1], q2 = mb[4], q3 = mb[5];

        v16h a, bv;
        a[0]  = (_Float16)p0.x; a[1]  = (_Float16)p0.y;
        a[2]  = (_Float16)p0.z; a[3]  = (_Float16)p0.w;
        a[4]  = (_Float16)p1.x; a[5]  = (_Float16)p1.y;
        a[6]  = (_Float16)p1.z; a[7]  = (_Float16)p1.w;
        a[8]  = (_Float16)p2.x; a[9]  = (_Float16)p2.y;
        a[10] = (_Float16)p2.z; a[11] = (_Float16)p2.w;
        a[12] = (_Float16)p3.x; a[13] = (_Float16)p3.y;
        a[14] = (_Float16)p3.z; a[15] = (_Float16)p3.w;

        bv[0]  = (_Float16)(q0.x * invL); bv[1]  = (_Float16)(q0.y * invL);
        bv[2]  = (_Float16)(q0.z * invL); bv[3]  = (_Float16)(q0.w * invL);
        bv[4]  = (_Float16)(q1.x * invL); bv[5]  = (_Float16)(q1.y * invL);
        bv[6]  = (_Float16)(q1.z * invL); bv[7]  = (_Float16)(q1.w * invL);
        bv[8]  = (_Float16)(q2.x * invL); bv[9]  = (_Float16)(q2.y * invL);
        bv[10] = (_Float16)(q2.z * invL); bv[11] = (_Float16)(q2.w * invL);
        bv[12] = (_Float16)(q3.x * invL); bv[13] = (_Float16)(q3.y * invL);
        bv[14] = (_Float16)(q3.z * invL); bv[15] = (_Float16)(q3.w * invL);

        acc = __builtin_amdgcn_wmma_f32_16x16x32_f16(
            /*neg_a=*/false, a, /*neg_b=*/false, bv,
            /*c_mod=*/(short)0, acc, /*reuse_a=*/false, /*reuse_b=*/false);
    }

    const float invD = 1.0f / (float)D_;
    if (m == 0) {                            // lanes 0 (rows 0-7) and 16 (rows 8-15)
        float s = 0.0f;
        float* yp = y1 + (size_t)b * L_ + l0 + g * 8;
        #pragma unroll
        for (int r = 0; r < 8; ++r) {
            float v = acc[r] * invD;
            yp[r] = v;
            s += v;
        }
        unsafeAtomicAdd(&msum[b], s);
    }
}

// ---------------------------------------------------------------------------
// Kernel 3: out[b,l,d] = alpha[l] + (y1[b,l] * msum[b] / L^2) * x[b,l,d]
// One block per (b,l) row; 192 threads = 192 float4 = full 768-wide row.
// ---------------------------------------------------------------------------
__global__ void __launch_bounds__(192)
finalize_kernel(const float* __restrict__ x, const float* __restrict__ alpha,
                const float* __restrict__ y1, const float* __restrict__ msum,
                float* __restrict__ out) {
    const size_t row = blockIdx.x;           // 0 .. B*L-1
    const int b = (int)(row >> 12);          // / 4096
    const int l = (int)(row & 4095);         // % 4096
    const float scale = y1[row] * msum[b] * (1.0f / ((float)L_ * (float)L_));
    const float av = alpha[l];
    const float4* xr = reinterpret_cast<const float4*>(x + row * D_);
    float4*       ow = reinterpret_cast<float4*>(out + row * D_);
    float4 v = xr[threadIdx.x];
    float4 o = make_float4(av + scale * v.x, av + scale * v.y,
                           av + scale * v.z, av + scale * v.w);
    ow[threadIdx.x] = o;
}

// ---------------------------------------------------------------------------
extern "C" void kernel_launch(void* const* d_in, const int* in_sizes, int n_in,
                              void* d_out, int out_size, void* d_ws, size_t ws_size,
                              hipStream_t stream) {
    (void)in_sizes; (void)n_in; (void)out_size; (void)ws_size;
    const float* x     = (const float*)d_in[0];   // [B, L, D]
    const float* alpha = (const float*)d_in[1];   // [L, 1]
    float* out = (float*)d_out;                   // [B, L, D]

    // Workspace layout: [meanSum B*D][msum B][y1 B*L]
    float* meanSum = (float*)d_ws;
    float* msum    = meanSum + B_ * D_;
    float* y1      = msum + B_;

    zero_ws_kernel<<<1, 256, 0, stream>>>(meanSum, B_ * D_ + B_);
    colsum_kernel<<<B_ * 64, 192, 0, stream>>>(x, meanSum);
    dot_wmma_kernel<<<dim3(L_ / 64, B_), 128, 0, stream>>>(x, meanSum, y1, msum);
    finalize_kernel<<<B_ * L_, 192, 0, stream>>>(x, alpha, y1, msum, out);
}
